// LRUBase_86741159510115
// MI455X (gfx1250) — compile-verified, matrix operand order checked by reading, not run on hardware
//
#include <hip/hip_runtime.h>
#include <hip/hip_bf16.h>
#include <cstdint>
#include <cstddef>

// Problem constants (match reference)
#define L_SEQ 16384
#define DIN   512
#define DH    1024
#define DOUT  512
#define CHUNK 128
#define NCHUNK (L_SEQ / CHUNK)   // 128

// LDS tiling: block tile 64(M) x 128(N), K-stage 64, row stride padded to 72
#define LDS_STRIDE 72            // bf16 elems; 144B = 36 banks -> conflict-free
#define A_ELEMS   (64 * LDS_STRIDE)    // 4608
#define W_ELEMS   (128 * LDS_STRIDE)   // 9216
#define BUF_ELEMS (A_ELEMS + W_ELEMS)  // 13824
#define BUF_BYTES (BUF_ELEMS * 2)      // 27648

typedef __bf16 bf16;
typedef __attribute__((ext_vector_type(16))) __bf16 v16bf;
typedef __attribute__((ext_vector_type(8)))  __bf16 v8bf;
typedef __attribute__((ext_vector_type(8)))  float  v8f;

// ---------- small helpers ----------
__device__ __forceinline__ bf16 f2bf(float f) {
  uint32_t u = __builtin_bit_cast(uint32_t, f);
  uint32_t r = u + 0x7FFFu + ((u >> 16) & 1u);   // round-to-nearest-even
  uint16_t h = (uint16_t)(r >> 16);
  return __builtin_bit_cast(bf16, h);
}
__device__ __forceinline__ float bf2f(bf16 x) {
  uint32_t u = ((uint32_t)__builtin_bit_cast(uint16_t, x)) << 16;
  return __builtin_bit_cast(float, u);
}

// Async global->LDS copy (ASYNCcnt) and its wait. ldsOff is a raw LDS byte
// offset (single static __shared__ array sits at group-segment offset 0).
// The generic pointer to the shared object is passed as an extra (unused)
// operand so the object ESCAPES into the asm; otherwise LLVM proves the
// non-escaping __shared__ array is never written and deletes the ds_loads.
__device__ __forceinline__ void async_b128(uint32_t ldsOff, uint64_t gaddr,
                                           const void* lds_obj) {
  asm volatile("global_load_async_to_lds_b128 %0, %1, off"
               :: "v"(ldsOff), "v"(gaddr), "v"(lds_obj) : "memory");
}
__device__ __forceinline__ void wait_async0() {
  asm volatile("s_wait_asynccnt 0" ::: "memory");
}

// Load one 16x32 bf16 WMMA operand fragment from an LDS tile with row stride
// LDS_STRIDE. lane l(<16): row=row0+l, K runs {k0..+7, k0+16..+23};
// lane l(>=16): row=row0+l-16, K runs {k0+8..+15, k0+24..+31}.
__device__ __forceinline__ v16bf load_frag(const bf16* __restrict__ p,
                                           int row0, int k0, int lane) {
  const int r = row0 + (lane & 15);
  const bf16* base = p + r * LDS_STRIDE + k0 + ((lane >> 4) << 3);
  v8bf lo = *(const v8bf*)(base);        // ds_load_b128
  v8bf hi = *(const v8bf*)(base + 16);   // ds_load_b128
  v16bf f;
#pragma unroll
  for (int i = 0; i < 8; ++i) { f[i] = lo[i]; f[i + 8] = hi[i]; }
  return f;
}

__device__ __forceinline__ v8f wmma_bf16(v16bf a, v16bf b, v8f c) {
  return __builtin_amdgcn_wmma_f32_16x16x32_bf16(false, a, false, b,
                                                 (short)0, c, false, false);
}

// ---------- shared GEMM core ----------
// 128 threads = 4 waves arranged 2M x 2N; wave tile 32x64 (acc 2x4) ->
// 12 ds_load_b128 per 8 WMMAs per K-step (1.5 loads/WMMA).
// Double-buffered LDS staging fed by async global->LDS copies.
__device__ __forceinline__ void gemm_block_accum(
    const bf16* __restrict__ A, const bf16* __restrict__ W, int K,
    bf16* smem, int mblk, int nblk, v8f (&acc)[2][4]) {
  const int tid  = threadIdx.x;
  const int lane = tid & 31;
  const int wid  = tid >> 5;          // 0..3
  const int wm = (wid >> 1) * 32;     // wave M offset in block tile
  const int wn = (wid & 1) * 64;      // wave N offset in block tile

  // Staging: each thread copies 16B per row-slab; 128 thr cover 16 rows/round.
  const int row  = tid >> 3;          // 0..15
  const int colb = (tid & 7) * 16;    // byte offset within 128B k-slab

  uint64_t ga[4];
  uint32_t la[4];
#pragma unroll
  for (int i = 0; i < 4; ++i) {
    ga[i] = (uint64_t)(uintptr_t)A + ((size_t)(mblk + row + 16 * i) * K) * 2 + colb;
    la[i] = (uint32_t)((row + 16 * i) * LDS_STRIDE * 2 + colb);
  }
  uint64_t gw[8];
  uint32_t lw[8];
#pragma unroll
  for (int i = 0; i < 8; ++i) {
    gw[i] = (uint64_t)(uintptr_t)W + ((size_t)(nblk + row + 16 * i) * K) * 2 + colb;
    lw[i] = (uint32_t)(A_ELEMS * 2 + (row + 16 * i) * LDS_STRIDE * 2 + colb);
  }

  const int KT = K >> 6;              // number of 64-wide K stages

  // prologue: fill buffer 0
  {
#pragma unroll
    for (int i = 0; i < 4; ++i) { async_b128(la[i], ga[i], smem); ga[i] += 128; }
#pragma unroll
    for (int i = 0; i < 8; ++i) { async_b128(lw[i], gw[i], smem); gw[i] += 128; }
  }
  wait_async0();
  __syncthreads();

  for (int kt = 0; kt < KT; ++kt) {
    const int cur = kt & 1;
    if (kt + 1 < KT) {
      const uint32_t bofs = (cur ^ 1) ? (uint32_t)BUF_BYTES : 0u;
#pragma unroll
      for (int i = 0; i < 4; ++i) { async_b128(la[i] + bofs, ga[i], smem); ga[i] += 128; }
#pragma unroll
      for (int i = 0; i < 8; ++i) { async_b128(lw[i] + bofs, gw[i], smem); gw[i] += 128; }
    }
    const bf16* sA = smem + cur * BUF_ELEMS;
    const bf16* sW = sA + A_ELEMS;
#pragma unroll
    for (int kk = 0; kk < 64; kk += 32) {
      v16bf a0 = load_frag(sA, wm,      kk, lane);
      v16bf a1 = load_frag(sA, wm + 16, kk, lane);
#pragma unroll
      for (int j = 0; j < 4; ++j) {
        v16bf b = load_frag(sW, wn + 16 * j, kk, lane);
        acc[0][j] = wmma_bf16(a0, b, acc[0][j]);
        acc[1][j] = wmma_bf16(a1, b, acc[1][j]);
      }
    }
    wait_async0();       // my async copies for the other buffer have landed
    __syncthreads();     // everyone's landed; everyone done reading `cur`
  }
}

// ---------- prep kernels ----------
__global__ void k_cvt_x(const float* __restrict__ x, bf16* __restrict__ xb, int n) {
  for (int i = blockIdx.x * blockDim.x + threadIdx.x; i < n;
       i += gridDim.x * blockDim.x)
    xb[i] = f2bf(x[i]);
}

__global__ void k_prep_lambda(const float* __restrict__ lamb,
                              float* __restrict__ lr, float* __restrict__ li) {
  int h = blockIdx.x * blockDim.x + threadIdx.x;
  if (h < DH) {
    float nu = expf(-expf(lamb[h]));       // |lambda|
    float th = expf(lamb[DH + h]);         // phase
    lr[h] = nu * cosf(th);
    li[h] = nu * sinf(th);
  }
}

__global__ void k_prep_b(const float* __restrict__ Bre, const float* __restrict__ Bim,
                         const float* __restrict__ gamma,
                         bf16* __restrict__ wbre, bf16* __restrict__ wbim) {
  int idx = blockIdx.x * blockDim.x + threadIdx.x;
  if (idx < DH * DIN) {
    float eg = expf(gamma[idx / DIN]);
    wbre[idx] = f2bf(Bre[idx] * eg);
    wbim[idx] = f2bf(Bim[idx] * eg);
  }
}

__global__ void k_prep_cd(const float* __restrict__ Cre, const float* __restrict__ Cim,
                          const float* __restrict__ Dm,
                          bf16* __restrict__ wcre, bf16* __restrict__ wcim,
                          bf16* __restrict__ wd) {
  int idx = blockIdx.x * blockDim.x + threadIdx.x;
  if (idx < DOUT * DH) {
    wcre[idx] = f2bf(Cre[idx]);
    wcim[idx] = f2bf(-Cim[idx]);   // fold the minus sign of -h_im@C_im^T
  }
  if (idx < DOUT * DIN) wd[idx] = f2bf(Dm[idx]);
}

// ---------- GEMM1: Bu = x @ W^T (bf16 out) ----------
__global__ void __launch_bounds__(128)
k_gemm_bu(const bf16* __restrict__ A, const bf16* __restrict__ W,
          bf16* __restrict__ out, int K, int ldo) {
  __shared__ __align__(16) bf16 smem[2 * BUF_ELEMS];
  const int lane = threadIdx.x & 31;
  const int wid  = threadIdx.x >> 5;
  const int mblk = blockIdx.y * 64;
  const int nblk = blockIdx.x * 128;
  v8f acc[2][4] = {};
  gemm_block_accum(A, W, K, smem, mblk, nblk, acc);

  const int m0 = mblk + (wid >> 1) * 32;
  const int n0 = nblk + (wid & 1) * 64;
  const int nn = lane & 15;
  const int mb = (lane >> 4) << 3;
#pragma unroll
  for (int i = 0; i < 2; ++i)
#pragma unroll
    for (int j = 0; j < 4; ++j) {
      int mBase = m0 + i * 16 + mb;
      int nCol  = n0 + j * 16 + nn;
#pragma unroll
      for (int r = 0; r < 8; ++r)
        out[(size_t)(mBase + r) * ldo + nCol] = f2bf(acc[i][j][r]);
    }
}

// ---------- scan kernels ----------
__global__ void k_scan1(const bf16* __restrict__ bure, const bf16* __restrict__ buim,
                        const float* __restrict__ lr, const float* __restrict__ li,
                        float* __restrict__ endre, float* __restrict__ endim) {
  int tid = blockIdx.x * blockDim.x + threadIdx.x;   // [NCHUNK*DH)
  int ch = tid & (DH - 1);
  int chunk = tid >> 10;                             // DH == 1024
  float a = lr[ch], b = li[ch];
  float hr = 0.f, hi = 0.f;
  size_t base = (size_t)chunk * CHUNK * DH + ch;
  for (int t = 0; t < CHUNK; ++t) {
    float br = bf2f(bure[base]), bi = bf2f(buim[base]);
    float nr = a * hr - b * hi + br;
    float ni = a * hi + b * hr + bi;
    hr = nr; hi = ni;
    base += DH;
  }
  endre[tid] = hr; endim[tid] = hi;
}

__global__ void k_combine(const float* __restrict__ endre, const float* __restrict__ endim,
                          const float* __restrict__ lr, const float* __restrict__ li,
                          float* __restrict__ hinre, float* __restrict__ hinim) {
  int ch = blockIdx.x * blockDim.x + threadIdx.x;
  if (ch >= DH) return;
  float a = lr[ch], b = li[ch];
  float pr = 1.f, pi = 0.f;                 // lambda^CHUNK
  for (int i = 0; i < CHUNK; ++i) {
    float nr = pr * a - pi * b;
    float ni = pr * b + pi * a;
    pr = nr; pi = ni;
  }
  float Hr = 0.f, Hi = 0.f;                 // state entering chunk j
  for (int j = 0; j < NCHUNK; ++j) {
    hinre[j * DH + ch] = Hr; hinim[j * DH + ch] = Hi;
    float er = endre[j * DH + ch], ei = endim[j * DH + ch];
    float nHr = pr * Hr - pi * Hi + er;
    float nHi = pr * Hi + pi * Hr + ei;
    Hr = nHr; Hi = nHi;
  }
}

__global__ void k_scan2(bf16* __restrict__ bure, bf16* __restrict__ buim,
                        const float* __restrict__ lr, const float* __restrict__ li,
                        const float* __restrict__ hinre, const float* __restrict__ hinim) {
  int tid = blockIdx.x * blockDim.x + threadIdx.x;
  int ch = tid & (DH - 1);
  int chunk = tid >> 10;
  float a = lr[ch], b = li[ch];
  float hr = hinre[tid], hi = hinim[tid];
  size_t base = (size_t)chunk * CHUNK * DH + ch;
  for (int t = 0; t < CHUNK; ++t) {
    float br = bf2f(bure[base]), bi = bf2f(buim[base]);
    float nr = a * hr - b * hi + br;
    float ni = a * hi + b * hr + bi;
    hr = nr; hi = ni;
    bure[base] = f2bf(hr);
    buim[base] = f2bf(hi);
    base += DH;
  }
}

// ---------- GEMM2 (fused): y = h_re@Cre^T + h_im@(-Cim)^T + x@D^T ----------
__global__ void __launch_bounds__(128)
k_gemm_out3(const bf16* __restrict__ A0, const bf16* __restrict__ W0, int K0,
            const bf16* __restrict__ A1, const bf16* __restrict__ W1, int K1,
            const bf16* __restrict__ A2, const bf16* __restrict__ W2, int K2,
            float* __restrict__ out, int ldo) {
  __shared__ __align__(16) bf16 smem[2 * BUF_ELEMS];
  const int lane = threadIdx.x & 31;
  const int wid  = threadIdx.x >> 5;
  const int mblk = blockIdx.y * 64;
  const int nblk = blockIdx.x * 128;
  v8f acc[2][4] = {};
  gemm_block_accum(A0, W0, K0, smem, mblk, nblk, acc);
  gemm_block_accum(A1, W1, K1, smem, mblk, nblk, acc);
  gemm_block_accum(A2, W2, K2, smem, mblk, nblk, acc);

  const int m0 = mblk + (wid >> 1) * 32;
  const int n0 = nblk + (wid & 1) * 64;
  const int nn = lane & 15;
  const int mb = (lane >> 4) << 3;
#pragma unroll
  for (int i = 0; i < 2; ++i)
#pragma unroll
    for (int j = 0; j < 4; ++j) {
      int mBase = m0 + i * 16 + mb;
      int nCol  = n0 + j * 16 + nn;
#pragma unroll
      for (int r = 0; r < 8; ++r)
        out[(size_t)(mBase + r) * ldo + nCol] = acc[i][j][r];
    }
}

// ---------- host launcher ----------
extern "C" void kernel_launch(void* const* d_in, const int* in_sizes, int n_in,
                              void* d_out, int out_size, void* d_ws, size_t ws_size,
                              hipStream_t stream) {
  const float* x     = (const float*)d_in[0];   // [L, DIN]
  const float* lamb  = (const float*)d_in[1];   // [2, DH]
  const float* gamma = (const float*)d_in[2];   // [DH]
  const float* Bre   = (const float*)d_in[3];   // [DH, DIN]
  const float* Bim   = (const float*)d_in[4];
  const float* Cre   = (const float*)d_in[5];   // [DOUT, DH]
  const float* Cim   = (const float*)d_in[6];
  const float* Dm    = (const float*)d_in[7];   // [DOUT, DIN]
  float* out = (float*)d_out;                   // [L, DOUT]

  char* w = (char*)d_ws;
  size_t off = 0;
  auto alloc = [&](size_t bytes) -> void* {
    void* p = w + off;
    off += (bytes + 255) & ~(size_t)255;
    return p;
  };
  bf16*  xbf   = (bf16*)alloc((size_t)L_SEQ * DIN * 2);   // 16 MB
  bf16*  wbre  = (bf16*)alloc((size_t)DH * DIN * 2);      // 1 MB
  bf16*  wbim  = (bf16*)alloc((size_t)DH * DIN * 2);
  bf16*  wcre  = (bf16*)alloc((size_t)DOUT * DH * 2);     // 1 MB
  bf16*  wcim  = (bf16*)alloc((size_t)DOUT * DH * 2);
  bf16*  wd    = (bf16*)alloc((size_t)DOUT * DIN * 2);
  float* lamre = (float*)alloc((size_t)DH * 4);
  float* lamim = (float*)alloc((size_t)DH * 4);
  bf16*  bure  = (bf16*)alloc((size_t)L_SEQ * DH * 2);    // 32 MB (Bu_re -> h_re)
  bf16*  buim  = (bf16*)alloc((size_t)L_SEQ * DH * 2);    // 32 MB (Bu_im -> h_im)
  float* endre = (float*)alloc((size_t)NCHUNK * DH * 4);
  float* endim = (float*)alloc((size_t)NCHUNK * DH * 4);
  float* hinre = (float*)alloc((size_t)NCHUNK * DH * 4);
  float* hinim = (float*)alloc((size_t)NCHUNK * DH * 4);

  // --- prep ---
  k_cvt_x<<<4096, 256, 0, stream>>>(x, xbf, L_SEQ * DIN);
  k_prep_lambda<<<(DH + 255) / 256, 256, 0, stream>>>(lamb, lamre, lamim);
  k_prep_b<<<(DH * DIN + 255) / 256, 256, 0, stream>>>(Bre, Bim, gamma, wbre, wbim);
  k_prep_cd<<<(DOUT * DH + 255) / 256, 256, 0, stream>>>(Cre, Cim, Dm, wcre, wcim, wd);

  // --- GEMM1: Bu_re, Bu_im (WMMA bf16, async-LDS staged) ---
  dim3 g1(DH / 128, L_SEQ / 64);
  k_gemm_bu<<<g1, 128, 0, stream>>>(xbf, wbre, bure, DIN, DH);
  k_gemm_bu<<<g1, 128, 0, stream>>>(xbf, wbim, buim, DIN, DH);

  // --- chunked complex scan h_t = lambda h_{t-1} + Bu_t ---
  k_scan1<<<(NCHUNK * DH) / 256, 256, 0, stream>>>(bure, buim, lamre, lamim, endre, endim);
  k_combine<<<DH / 256, 256, 0, stream>>>(endre, endim, lamre, lamim, hinre, hinim);
  k_scan2<<<(NCHUNK * DH) / 256, 256, 0, stream>>>(bure, buim, lamre, lamim, hinre, hinim);

  // --- GEMM2 fused: y = h_re@Cre^T + h_im@(-Cim)^T + x@D^T ---
  dim3 g2(DOUT / 128, L_SEQ / 64);
  k_gemm_out3<<<g2, 128, 0, stream>>>(bure, wcre, DH,
                                      buim, wcim, DH,
                                      xbf,  wd,   DIN,
                                      out, DOUT);
}
